// Voxel_1924145348768
// MI455X (gfx1250) — compile-verified, matrix operand order checked by reading, not run on hardware
//
#include <hip/hip_runtime.h>

// ---------------------------------------------------------------------------
// Voxel cosine-basis contraction as a single WMMA GEMM:
//   OUT(16 x 32768) = VOX_f16(16 x 8192) x E(8192 x 32768)
// E[k,q] = ct[t]*ch[h]*cw[w] generated on the fly per lane (never stored).
// A-operand streamed into LDS with the Tensor Data Mover (double-buffered).
// Inner loop: A-fragment loads issued 2 chunks ahead of their WMMA (loads
// precede the consuming wmma in program order, forcing 3 live A buffers),
// so the pre-WMMA wait is a partial s_wait_dscnt instead of dscnt==0.
// ---------------------------------------------------------------------------

typedef __attribute__((ext_vector_type(16))) _Float16 v16h;
typedef __attribute__((ext_vector_type(8)))  float    v8f;
typedef __attribute__((ext_vector_type(4)))  unsigned int u32x4;
typedef __attribute__((ext_vector_type(8)))  int      i32x8;
typedef __attribute__((ext_vector_type(4)))  int      i32x4;

#define PI_F 3.14159265358979323846f

constexpr int Cc     = 16;              // channels (WMMA M)
constexpr int Tt     = 8;
constexpr int Hh     = 32;
constexpr int Ww     = 32;
constexpr int KTOT   = Tt * Hh * Ww;    // 8192
constexpr int NCHUNK = KTOT / 32;       // 256 K-chunks of 32
constexpr int GRPCH  = 16;              // chunks per LDS group (16 KB)
constexpr int NGRP   = NCHUNK / GRPCH;  // 16 groups
constexpr int GRPBYTES = GRPCH * 32 * 16 * 2;   // 16384 bytes per group
constexpr int HQW    = 128 * 128;       // 16384 queries per batch image

// ---------------------------------------------------------------------------
// Pre-pass: convert voxel f32 -> f16, swizzled into per-lane A-fragment order.
// Fragment layout for V_WMMA_F32_16X16X32_F16 A-matrix (16x32, 16-bit):
//   lane L (0..31): row M = L&15, half = L>>4
//   element e (0..15): K = (e<8 ? 0 : 16) + half*8 + (e&7)
// Output layout: frag[m][lane][e] halves, m = K-chunk index (0..255).
// ---------------------------------------------------------------------------
__global__ void convert_voxel_kernel(const float* __restrict__ vox,
                                     _Float16* __restrict__ out) {
    int i = blockIdx.x * blockDim.x + threadIdx.x;   // 0 .. 131071
    int m  = i >> 9;          // chunk
    int l  = (i >> 4) & 31;   // lane
    int e  = i & 15;          // element in fragment
    int c  = l & 15;          // channel row
    int hl = l >> 4;          // lane half
    int K  = ((e & 8) ? 16 : 0) + hl * 8 + (e & 7);  // K within chunk
    float v = vox[c * KTOT + m * 32 + K];
    out[i] = (_Float16)v;
}

// ---------------------------------------------------------------------------
// TDM: DMA-copy 16 KB (4096 x 4B, one row) global -> LDS. Issued by one wave;
// EXEC is ignored for tensor ops, completion tracked with TENSORcnt.
// ---------------------------------------------------------------------------
#if __has_builtin(__builtin_amdgcn_tensor_load_to_lds)
#define HAVE_TDM 1
__device__ __forceinline__ void tdm_copy16k(const void* gsrc, unsigned lds_byte) {
    unsigned long long ga = (unsigned long long)(uintptr_t)gsrc;
    u32x4 g0;
    g0[0] = 1u;                                   // count = 1 (valid descriptor)
    g0[1] = lds_byte;                             // lds_addr  (bits 63:32)
    g0[2] = (unsigned)(ga & 0xffffffffull);       // global_addr lo (bits 95:64)
    g0[3] = (unsigned)((ga >> 32) & 0x1ffffffull) // global_addr hi (bits 120:96)
          | (2u << 30);                           // type = 2 "image" (127:126)
    i32x8 g1 = {0, 0, 0, 0, 0, 0, 0, 0};
    g1[0] = (2 << 16);                            // data_size = 4 bytes
    g1[1] = (int)(4096u << 16);                   // tensor_dim0 = 4096 (lo16 @63:48)
    g1[2] = (1 << 16);                            // dim0 hi16=0 | tensor_dim1 = 1
    g1[3] = (int)(4096u << 16);                   // dim1 hi16=0 | tile_dim0 = 4096
    g1[4] = 1;                                    // tile_dim1 = 1, tile_dim2 = 0
    g1[5] = 4096;                                 // tensor_dim0_stride = 4096
    g1[6] = (int)(4096u << 16);                   // tensor_dim1_stride lo16
    g1[7] = 0;
    i32x4 g2 = {0, 0, 0, 0};
    i32x4 g3 = {0, 0, 0, 0};
#if __has_include(<hip/amd_detail/amd_gfx1250_TDM.h>)
    i32x8 g4 = {0, 0, 0, 0, 0, 0, 0, 0};          // therock clang-23: 6-arg form
    __builtin_amdgcn_tensor_load_to_lds(g0, g1, g2, g3, g4, 0);
#else
    __builtin_amdgcn_tensor_load_to_lds(g0, g1, g2, g3, 0);  // ROCm 7.2: 5-arg
#endif
}
#endif

// ---------------------------------------------------------------------------
// Main kernel: 128 threads = 4 wave32; each wave computes a 16-query N-tile.
// grid = 512 blocks * 64 queries/block = 32768 queries exactly.
// ---------------------------------------------------------------------------
__global__ void __launch_bounds__(128)
voxel_wmma_kernel(const float* __restrict__ grid,
                  const _Float16* __restrict__ voxh,
                  const float* __restrict__ pose_r,
                  const float* __restrict__ pose_t,
                  float* __restrict__ out) {
    extern __shared__ char smem_raw[];                 // 2 x 16 KB double buffer
    _Float16* smem = (_Float16*)smem_raw;
    const char* gbytes = (const char*)voxh;

    const int tid  = threadIdx.x;
    const int wave = tid >> 5;
    const int lane = tid & 31;
    const int hv   = lane >> 4;      // lane half
    const int nq   = lane & 15;      // query slot in this wave's N-tile
    const int q    = blockIdx.x * 64 + wave * 16 + nq;

    // --- pose: row-normalize 3x3 (F.normalize dim=1), then g = grid @ M + t ---
    float mat[3][3];
#pragma unroll
    for (int r = 0; r < 3; ++r) {
        float a = pose_r[r * 3 + 0];
        float b = pose_r[r * 3 + 1];
        float c = pose_r[r * 3 + 2];
        float nrm = sqrtf(a * a + b * b + c * c);
        float inv = 1.0f / fmaxf(nrm, 1e-12f);
        mat[r][0] = a * inv; mat[r][1] = b * inv; mat[r][2] = c * inv;
    }
    float g0 = grid[q * 3 + 0], g1 = grid[q * 3 + 1], g2 = grid[q * 3 + 2];
    float gx = g0 * mat[0][0] + g1 * mat[1][0] + g2 * mat[2][0] + pose_t[0];
    float gy = g0 * mat[0][1] + g1 * mat[1][1] + g2 * mat[2][1] + pose_t[1];
    float gz = g0 * mat[0][2] + g1 * mat[1][2] + g2 * mat[2][2] + pose_t[2];
    gx = (gx + 1.0f) * 0.5f;
    gy = (gy + 1.0f) * 0.5f;
    gz = (gz + 1.0f) * 0.5f;

    // --- per-lane B base: cw at this lane's 16 K-slots (mirrors A layout,
    //     lane holds column N = nq). Fixed for the whole K loop. ---
    v16h bb;
#pragma unroll
    for (int e = 0; e < 16; ++e) {
        int K = ((e & 8) ? 16 : 0) + hv * 8 + (e & 7);   // w index within chunk
        bb[e] = (_Float16)__cosf(gz * (PI_F * (float)K + PI_F * 0.5f));
    }

    v8f acc = {};

#if defined(HAVE_TDM)
    const unsigned lds_base = (unsigned)(uintptr_t)smem_raw;  // LDS byte offset

    // prologue: TDM-copy group 0 into buffer 0
    if (wave == 0) {
        tdm_copy16k(gbytes, lds_base);
        __builtin_amdgcn_s_wait_tensorcnt(0);
    }
    __syncthreads();
#else
    {   // fallback: cooperative copy of group 0 (8 uint4 per thread)
        uint4* sdst = (uint4*)smem_raw;
        const uint4* gsrc = (const uint4*)voxh;
#pragma unroll
        for (int j = 0; j < 8; ++j) sdst[j * 128 + tid] = gsrc[j * 128 + tid];
    }
    __syncthreads();
#endif

    // --- K loop: 16 groups x 16 chunks, double-buffered LDS ---
    for (int grp = 0; grp < NGRP; ++grp) {
        const int p = grp & 1;

#if defined(HAVE_TDM)
        // kick off next group's DMA into the other buffer; overlaps the WMMAs
        if (wave == 0 && grp + 1 < NGRP)
            tdm_copy16k(gbytes + (grp + 1) * GRPBYTES,
                        lds_base + (p ^ 1) * GRPBYTES);
#else
        uint4 nxt[8];
        const uint4* gsrc = (const uint4*)voxh;
        if (grp + 1 < NGRP) {
#pragma unroll
            for (int j = 0; j < 8; ++j)
                nxt[j] = gsrc[(grp + 1) * 1024 + j * 128 + tid];
        }
#endif

        // Precompute the 16 per-chunk scalars s = ct[t]*ch[h], packed into one
        // v16h, OFF the WMMA critical path. t = grp>>1 is constant per group.
        // The empty asm is an optimization barrier: forces materialization and
        // stops the compiler from sinking the cos chain back into the WMMA loop.
        v16h sv;
        {
            float ctv = __cosf(gx * (PI_F * (float)(grp >> 1) + PI_F * 0.5f));
#pragma unroll
            for (int cc = 0; cc < GRPCH; ++cc) {
                int h = ((grp & 1) << 4) + cc;
                float chv = __cosf(gy * (PI_F * (float)h + PI_F * 0.5f));
                sv[cc] = (_Float16)(ctv * chv);
            }
            asm volatile("" : "+v"(sv));
        }

        // WMMA loop with a真 2-deep A prefetch: the ds_load of a(cc+2) is
        // issued BEFORE wmma(a(cc)) in program order, so abuf[0..2] have
        // overlapping lifetimes (allocator cannot coalesce them) and the
        // pre-WMMA wait becomes a partial dscnt wait (2 load-pairs in flight).
        const _Float16* abase = smem + p * (GRPCH * 32 * 16) + lane * 16;
        v16h abuf[3];
        abuf[0] = *(const v16h*)(abase + 0 * (32 * 16));
        abuf[1] = *(const v16h*)(abase + 1 * (32 * 16));
#pragma unroll
        for (int cc = 0; cc < GRPCH; ++cc) {
            if (cc + 2 < GRPCH)
                abuf[(cc + 2) % 3] = *(const v16h*)(abase + (cc + 2) * (32 * 16));
            v16h bfrag = bb * sv[cc];                             // v_pk_mul_f16
            acc = __builtin_amdgcn_wmma_f32_16x16x32_f16(
                false, abuf[cc % 3], false, bfrag, (short)0, acc, false, false);
        }

#if defined(HAVE_TDM)
        if (wave == 0 && grp + 1 < NGRP)
            __builtin_amdgcn_s_wait_tensorcnt(0);   // next group's data landed
#else
        if (grp + 1 < NGRP) {
            uint4* sdst = (uint4*)smem_raw;
#pragma unroll
            for (int j = 0; j < 8; ++j)
                sdst[(p ^ 1) * 1024 + j * 128 + tid] = nxt[j];
        }
#endif
        __syncthreads();
    }

    // --- write out: D layout => lane holds D[M = r + hv*8, N = nq], M = channel.
    //     out[b][c][i][j], rem = i*128 + j ---
    int b   = q >> 14;
    int rem = q & (HQW - 1);
    float* obase = out + b * (Cc * HQW) + rem;
#pragma unroll
    for (int r = 0; r < 8; ++r) {
        int c = hv * 8 + r;
        obase[c * HQW] = acc[r];
    }
}

// ---------------------------------------------------------------------------
extern "C" void kernel_launch(void* const* d_in, const int* in_sizes, int n_in,
                              void* d_out, int out_size, void* d_ws, size_t ws_size,
                              hipStream_t stream) {
    const float* grid   = (const float*)d_in[0];  // (2,128,128,3)
    const float* voxel  = (const float*)d_in[1];  // (16,8,32,32)
    const float* pose_r = (const float*)d_in[2];  // (3,3)
    const float* pose_t = (const float*)d_in[3];  // (3,)
    // d_in[4..6] = t_map/h_map/w_map: recomputed in-kernel as pi*(k+0.5)

    _Float16* voxh = (_Float16*)d_ws;             // 131072 halves = 256 KB

    // 1) convert + swizzle voxel into WMMA A-fragment order
    convert_voxel_kernel<<<512, 256, 0, stream>>>(voxel, voxh);

    // 2) main WMMA contraction: 512 blocks x 128 threads, 32 KB dynamic LDS
    voxel_wmma_kernel<<<512, 128, 2 * GRPBYTES, stream>>>(
        grid, voxh, pose_r, pose_t, (float*)d_out);
}
